// MultiHeadedAttention_8108898255193
// MI455X (gfx1250) — matrix operand order repaired and checked by reading, not compiled, not run on hardware
//
#include <hip/hip_runtime.h>

#define DEV __device__ __forceinline__

typedef __attribute__((ext_vector_type(16))) __bf16        v16bf;
typedef __attribute__((ext_vector_type(8)))  float         v8f;
typedef __attribute__((ext_vector_type(8)))  unsigned short v8u;
typedef __attribute__((ext_vector_type(16))) unsigned short v16u;

static constexpr int Bb = 2, Ss = 2048, Dd = 1024, Hh = 16, DKk = 64;

DEV unsigned short f2bf(float f) {
  unsigned u = __builtin_bit_cast(unsigned, f);
  u += 0x7FFFu + ((u >> 16) & 1u);          // round-to-nearest-even
  return (unsigned short)(u >> 16);
}
DEV v16bf as_bf(v16u v) { return __builtin_bit_cast(v16bf, v); }

// ---- DPP16 16-lane butterfly reductions (stay inside each 16-lane half) ----
// quad_perm xor1 = 0xB1, quad_perm xor2 = 0x4E, row_half_mirror = 0x141, row_mirror = 0x140
template <int CTRL>
DEV float dpp_f32(float x) {
  int xi = __builtin_bit_cast(int, x);
  int r  = __builtin_amdgcn_update_dpp(xi, xi, CTRL, 0xF, 0xF, true);
  return __builtin_bit_cast(float, r);
}
DEV float red_max16(float x) {
  x = fmaxf(x, dpp_f32<0xB1>(x));
  x = fmaxf(x, dpp_f32<0x4E>(x));
  x = fmaxf(x, dpp_f32<0x141>(x));
  x = fmaxf(x, dpp_f32<0x140>(x));
  return x;
}
DEV float red_sum16(float x) {
  x += dpp_f32<0xB1>(x);
  x += dpp_f32<0x4E>(x);
  x += dpp_f32<0x141>(x);
  x += dpp_f32<0x140>(x);
  return x;
}

// ---------------- stage 1: fp32 -> bf16 converts ----------------
__global__ void cvt_bf16(const float* __restrict__ in, unsigned short* __restrict__ out, int n) {
  int i = blockIdx.x * blockDim.x + threadIdx.x;
  if (i < n) out[i] = f2bf(in[i]);
}

// W[k][n] (1024x1024 fp32) -> WT[n][k] bf16
__global__ void transpose_bf16(const float* __restrict__ W, unsigned short* __restrict__ WT) {
  int i = blockIdx.x * blockDim.x + threadIdx.x;   // over 1M outputs
  int n = i >> 10, k = i & 1023;
  WT[i] = f2bf(W[((size_t)k << 10) + n]);
}

// ---------------- stage 2/4: WMMA GEMM, C = A[4096x1024] @ W + bias ----------------
// One wave computes a 32x32 output tile (2x2 WMMA tiles): 8 b128 loads feed 4 WMMAs.
// MODE 0: bf16 row-major   MODE 1: bf16 per-head-transposed vT[b][h][d][s]   MODE 2: fp32
template <int MODE>
DEV void store_tile(void* __restrict__ outp, int m_base, int n_col, const v8f& acc, float bb) {
#pragma unroll
  for (int r = 0; r < 8; ++r) {
    const int m = m_base + r;
    const float val = acc[r] + bb;
    if (MODE == 0) {
      ((unsigned short*)outp)[((size_t)m << 10) + n_col] = f2bf(val);
    } else if (MODE == 1) {
      const int hh = n_col >> 6, d = n_col & 63;
      const int bi = m >> 11, s = m & 2047;
      ((unsigned short*)outp)[((((size_t)bi * Hh + hh) * DKk) + d) * Ss + s] = f2bf(val);
    } else {
      ((float*)outp)[((size_t)m << 10) + n_col] = val;
    }
  }
}

template <int MODE>
__global__ void gemm_wmma_bf16(const unsigned short* __restrict__ A,
                               const unsigned short* __restrict__ WT,
                               const float* __restrict__ bias,
                               void* __restrict__ outp) {
  const int wid  = blockIdx.x * (blockDim.x >> 5) + (threadIdx.x >> 5);
  const int lane = threadIdx.x & 31;
  const int lh = lane & 15, hi = lane >> 4;
  const int m0 = (wid >> 5) << 5;                  // N=1024 -> 32 column supertiles
  const int n0 = (wid & 31) << 5;

  const unsigned short* arow0 = A  + ((size_t)(m0 + lh) << 10);
  const unsigned short* arow1 = arow0 + ((size_t)16 << 10);
  const unsigned short* brow0 = WT + ((size_t)(n0 + lh) << 10) + hi * 8;
  const unsigned short* brow1 = brow0 + ((size_t)16 << 10);

  v8f acc00 = {}, acc01 = {}, acc10 = {}, acc11 = {};
  for (int k0 = 0; k0 < 1024; k0 += 32) {
    v16u a0, a1, b0, b1;
    *(v8u*)&a0       = *(const v8u*)(arow0 + k0 + hi * 8);
    *((v8u*)&a0 + 1) = *(const v8u*)(arow0 + k0 + 16 + hi * 8);
    *(v8u*)&a1       = *(const v8u*)(arow1 + k0 + hi * 8);
    *((v8u*)&a1 + 1) = *(const v8u*)(arow1 + k0 + 16 + hi * 8);
    *(v8u*)&b0       = *(const v8u*)(brow0 + k0);
    *((v8u*)&b0 + 1) = *(const v8u*)(brow0 + k0 + 16);
    *(v8u*)&b1       = *(const v8u*)(brow1 + k0);
    *((v8u*)&b1 + 1) = *(const v8u*)(brow1 + k0 + 16);
    const v16bf af0 = as_bf(a0), af1 = as_bf(a1), bf0 = as_bf(b0), bf1 = as_bf(b1);
    acc00 = __builtin_amdgcn_wmma_f32_16x16x32_bf16(false, af0, false, bf0, (short)0, acc00, false, false);
    acc01 = __builtin_amdgcn_wmma_f32_16x16x32_bf16(false, af0, false, bf1, (short)0, acc01, false, false);
    acc10 = __builtin_amdgcn_wmma_f32_16x16x32_bf16(false, af1, false, bf0, (short)0, acc10, false, false);
    acc11 = __builtin_amdgcn_wmma_f32_16x16x32_bf16(false, af1, false, bf1, (short)0, acc11, false, false);
  }
  const float bb0 = bias[n0 + lh];
  const float bb1 = bias[n0 + 16 + lh];
  store_tile<MODE>(outp, m0 + hi * 8,      n0 + lh,      acc00, bb0);
  store_tile<MODE>(outp, m0 + hi * 8,      n0 + 16 + lh, acc01, bb1);
  store_tile<MODE>(outp, m0 + 16 + hi * 8, n0 + lh,      acc10, bb0);
  store_tile<MODE>(outp, m0 + 16 + hi * 8, n0 + 16 + lh, acc11, bb1);
}

// ---------------- stage 3: flash attention, one wave per (b,h, 16 query rows) ----------------
__global__ void flash_attn_wmma(const unsigned short* __restrict__ qb,
                                const unsigned short* __restrict__ kb,
                                const unsigned short* __restrict__ vT,
                                unsigned short* __restrict__ xb) {
  const int wv   = threadIdx.x >> 5;
  const int wid  = blockIdx.x * (blockDim.x >> 5) + wv;
  const int lane = threadIdx.x & 31;
  const int lh = lane & 15, hi = lane >> 4;

  const int qblk = wid & 127;          // S/16
  const int h    = (wid >> 7) & 15;
  const int b    = wid >> 11;

  __shared__ unsigned short psh[4][16 * 40];     // 16 rows x 32 bf16, stride 40 (16B aligned)
  unsigned short* sh = psh[wv];

  // Q fragments (A layout) for d-chunks 0..31 and 32..63 -- loaded once
  const unsigned short* qrow = qb + ((size_t)(b * Ss + qblk * 16 + lh) << 10) + h * DKk;
  v16u a0u, a1u;
  *(v8u*)&a0u       = *(const v8u*)(qrow + hi * 16);
  *((v8u*)&a0u + 1) = *(const v8u*)(qrow + hi * 16 + 8);
  *(v8u*)&a1u       = *(const v8u*)(qrow + 32 + hi * 16);
  *((v8u*)&a1u + 1) = *(const v8u*)(qrow + 32 + hi * 16 + 8);
  const v16bf aq0 = as_bf(a0u), aq1 = as_bf(a1u);

  v8f o0 = {}, o1 = {}, o2 = {}, o3 = {};
  float mrow[8], lrow[8];
#pragma unroll
  for (int r = 0; r < 8; ++r) { mrow[r] = -1e30f; lrow[r] = 0.0f; }

  const unsigned short* kbase = kb + ((size_t)b * Ss << 10) + h * DKk;
  const unsigned short* vbase = vT + (((size_t)b * Hh + h) * DKk) * Ss;

  for (int j = 0; j < Ss / 32; ++j) {
    const int kp = j * 32;
    // prefetch next key/value tiles (L2-resident stream; lowers to global_prefetch_b8)
    if (j + 1 < Ss / 32) {
      __builtin_prefetch(kbase + ((size_t)(kp + 32 + lh) << 10) + hi * 16, 0, 0);
      __builtin_prefetch(vbase + (size_t)lh * Ss + kp + 32 + hi * 16, 0, 0);
    }
    // ---- scores S = (Q K^T) * 1/sqrt(DK), two 16-key groups ----
    v8f s0 = {}, s1 = {};
    {
      const unsigned short* k0r = kbase + ((size_t)(kp + lh) << 10);
      const unsigned short* k1r = kbase + ((size_t)(kp + 16 + lh) << 10);
      v16u b00, b01, b10, b11;
      *(v8u*)&b00       = *(const v8u*)(k0r + hi * 16);
      *((v8u*)&b00 + 1) = *(const v8u*)(k0r + hi * 16 + 8);
      *(v8u*)&b01       = *(const v8u*)(k0r + 32 + hi * 16);
      *((v8u*)&b01 + 1) = *(const v8u*)(k0r + 32 + hi * 16 + 8);
      *(v8u*)&b10       = *(const v8u*)(k1r + hi * 16);
      *((v8u*)&b10 + 1) = *(const v8u*)(k1r + hi * 16 + 8);
      *(v8u*)&b11       = *(const v8u*)(k1r + 32 + hi * 16);
      *((v8u*)&b11 + 1) = *(const v8u*)(k1r + 32 + hi * 16 + 8);
      s0 = __builtin_amdgcn_wmma_f32_16x16x32_bf16(false, aq0, false, as_bf(b00), (short)0, s0, false, false);
      s0 = __builtin_amdgcn_wmma_f32_16x16x32_bf16(false, aq1, false, as_bf(b01), (short)0, s0, false, false);
      s1 = __builtin_amdgcn_wmma_f32_16x16x32_bf16(false, aq0, false, as_bf(b10), (short)0, s1, false, false);
      s1 = __builtin_amdgcn_wmma_f32_16x16x32_bf16(false, aq1, false, as_bf(b11), (short)0, s1, false, false);
    }
    // ---- online softmax: DPP butterfly reductions, no LDS, no waits ----
#pragma unroll
    for (int r = 0; r < 8; ++r) {
      float x0 = s0[r] * 0.125f, x1 = s1[r] * 0.125f;
      const float mx    = red_max16(fmaxf(x0, x1));
      const float mnew  = fmaxf(mrow[r], mx);
      const float alpha = __expf(mrow[r] - mnew);
      mrow[r] = mnew;
      const float p0 = __expf(x0 - mnew);
      const float p1 = __expf(x1 - mnew);
      const float rs = red_sum16(p0 + p1);
      lrow[r] = lrow[r] * alpha + rs;
      o0[r] *= alpha; o1[r] *= alpha; o2[r] *= alpha; o3[r] *= alpha;
      const int row = hi * 8 + r;
      sh[row * 40 + lh]      = f2bf(p0);     // D-layout -> LDS tile (16x32 bf16)
      sh[row * 40 + 16 + lh] = f2bf(p1);
    }
    asm volatile("s_wait_dscnt 0" ::: "memory");   // in-wave LDS store->load ordering
    // ---- re-read P in A-fragment layout ----
    v16u pu;
    *(v8u*)&pu       = *(const v8u*)(sh + lh * 40 + hi * 16);
    *((v8u*)&pu + 1) = *(const v8u*)(sh + lh * 40 + hi * 16 + 8);
    const v16bf ap = as_bf(pu);
    // ---- O += P @ V (V pre-transposed: contiguous key positions per d-column) ----
#pragma unroll
    for (int f = 0; f < 4; ++f) {
      const unsigned short* vrow = vbase + (size_t)(f * 16 + lh) * Ss + kp + hi * 16;
      v16u bv;
      *(v8u*)&bv       = *(const v8u*)(vrow);
      *((v8u*)&bv + 1) = *(const v8u*)(vrow + 8);
      v8f* op = (f == 0) ? &o0 : (f == 1) ? &o1 : (f == 2) ? &o2 : &o3;
      *op = __builtin_amdgcn_wmma_f32_16x16x32_bf16(false, ap, false, as_bf(bv), (short)0, *op, false, false);
    }
  }
  // ---- normalize and store bf16 into [B,S,D] ----
#pragma unroll
  for (int r = 0; r < 8; ++r) {
    const float inv = 1.0f / lrow[r];
    const int srow = qblk * 16 + hi * 8 + r;
    unsigned short* orow = xb + ((size_t)(b * Ss + srow) << 10) + h * DKk;
    orow[lh]      = f2bf(o0[r] * inv);
    orow[16 + lh] = f2bf(o1[r] * inv);
    orow[32 + lh] = f2bf(o2[r] * inv);
    orow[48 + lh] = f2bf(o3[r] * inv);
  }
}

// ---------------- host ----------------
extern "C" void kernel_launch(void* const* d_in, const int* in_sizes, int n_in,
                              void* d_out, int out_size, void* d_ws, size_t ws_size,
                              hipStream_t stream) {
  const float* query = (const float*)d_in[0];
  const float* key   = (const float*)d_in[1];
  const float* value = (const float*)d_in[2];
  const float* Wq = (const float*)d_in[3];  const float* bq = (const float*)d_in[4];
  const float* Wk = (const float*)d_in[5];  const float* bk = (const float*)d_in[6];
  const float* Wv = (const float*)d_in[7];  const float* bv = (const float*)d_in[8];
  const float* Wo = (const float*)d_in[9];  const float* bo = (const float*)d_in[10];

  const size_t NE = (size_t)Bb * Ss * Dd;   // 4,194,304
  const size_t WE = (size_t)Dd * Dd;        // 1,048,576

  char* ws = (char*)d_ws;
  size_t off = 0;
  auto alloc = [&](size_t bytes) -> void* {
    void* p = ws + off;
    off = (off + bytes + 255) & ~(size_t)255;
    return p;
  };
  unsigned short* qin = (unsigned short*)alloc(NE * 2);
  unsigned short* kin = (unsigned short*)alloc(NE * 2);
  unsigned short* vin = (unsigned short*)alloc(NE * 2);
  unsigned short* qbb = (unsigned short*)alloc(NE * 2);
  unsigned short* kbb = (unsigned short*)alloc(NE * 2);
  unsigned short* vTb = (unsigned short*)alloc(NE * 2);
  unsigned short* xbb = (unsigned short*)alloc(NE * 2);
  unsigned short* WqT = (unsigned short*)alloc(WE * 2);
  unsigned short* WkT = (unsigned short*)alloc(WE * 2);
  unsigned short* WvT = (unsigned short*)alloc(WE * 2);
  unsigned short* WoT = (unsigned short*)alloc(WE * 2);

  // stage 1: converts
  cvt_bf16<<<dim3((unsigned)(NE / 256)), 256, 0, stream>>>(query, qin, (int)NE);
  cvt_bf16<<<dim3((unsigned)(NE / 256)), 256, 0, stream>>>(key,   kin, (int)NE);
  cvt_bf16<<<dim3((unsigned)(NE / 256)), 256, 0, stream>>>(value, vin, (int)NE);
  transpose_bf16<<<dim3((unsigned)(WE / 256)), 256, 0, stream>>>(Wq, WqT);
  transpose_bf16<<<dim3((unsigned)(WE / 256)), 256, 0, stream>>>(Wk, WkT);
  transpose_bf16<<<dim3((unsigned)(WE / 256)), 256, 0, stream>>>(Wv, WvT);
  transpose_bf16<<<dim3((unsigned)(WE / 256)), 256, 0, stream>>>(Wo, WoT);

  // stage 2: QKV projections (4096 x 32x32-tile waves -> 1024 blocks of 4 waves)
  const dim3 ggrid(1024), gblk(128);
  gemm_wmma_bf16<0><<<ggrid, gblk, 0, stream>>>(qin, WqT, bq, qbb);
  gemm_wmma_bf16<0><<<ggrid, gblk, 0, stream>>>(kin, WkT, bk, kbb);
  gemm_wmma_bf16<1><<<ggrid, gblk, 0, stream>>>(vin, WvT, bv, vTb);

  // stage 3: attention (B*H*(S/16) = 4096 waves)
  flash_attn_wmma<<<dim3(1024), dim3(128), 0, stream>>>(qbb, kbb, vTb, xbb);

  // stage 4: output projection, fp32 to d_out
  gemm_wmma_bf16<2><<<ggrid, gblk, 0, stream>>>(xbb, WoT, bo, d_out);
}
